// EntityAwareBertAttention_13030930776492
// MI455X (gfx1250) — compile-verified, matrix operand order checked by reading, not run on hardware
//
#include <hip/hip_runtime.h>
#include <hip/hip_bf16.h>
#include <cstdint>

// ---------------------------------------------------------------------------
// Types for CDNA5 WMMA (wave32): 16x16x32 f16 -> f32
// ---------------------------------------------------------------------------
typedef __attribute__((ext_vector_type(16))) _Float16 v16h;
typedef __attribute__((ext_vector_type(8)))  _Float16 v8h;
typedef __attribute__((ext_vector_type(8)))  float    v8f;

#define HD     1024
#define NHEAD  16
#define DHEAD  64
#define BATCH  4
#define TW     1024
#define TE     128
#define TT     1152   // TW + TE

// ---------------------------------------------------------------------------
// CDNA5 async global->LDS staging (ASYNCcnt-tracked, bypasses VGPRs).
// LDS byte offset = low 32 bits of the generic shared-pointer address
// (flat->LDS mapping truncates to addr[31:0], hardware adds the wave base).
// ---------------------------------------------------------------------------
__device__ __forceinline__ unsigned lds_off(const void* p) {
  return (unsigned)(unsigned long long)p;
}

__device__ __forceinline__ void async_ld_b128(unsigned lds_byte_off, const void* gaddr) {
  asm volatile("global_load_async_to_lds_b128 %0, %1, off"
               :: "v"(lds_byte_off), "v"(gaddr) : "memory");
}

__device__ __forceinline__ void wait_async_le2() {
  asm volatile("s_wait_asynccnt 0x2" ::: "memory");
}
__device__ __forceinline__ void wait_async_le4() {
  asm volatile("s_wait_asynccnt 0x4" ::: "memory");
}
__device__ __forceinline__ void wait_async_0() {
  asm volatile("s_wait_asynccnt 0x0" ::: "memory");
}

// ---------------------------------------------------------------------------
// Tensor Data Mover: one instruction DMAs a 2D tile global->LDS, tracked by
// TENSORcnt. This toolchain exposes the 6-arg builtin
//   (uint32x4 g0, int32x8 g1, int32x4 g2, int32x4 g3, int32x8 g4, i32 cpol)
// We describe: data_size=2B, tile 32(halves) x 128(rows), row stride K,
// pad_enable with pad_interval=16 DWORDs / pad_amount=4 DWORDs, which
// reproduces the LDA=40-half padded LDS layout used by the fragment loads.
// Guarded by __has_builtin; falls back to per-lane async loads otherwise.
// ---------------------------------------------------------------------------
#if defined(__HIP_DEVICE_COMPILE__) && __has_builtin(__builtin_amdgcn_tensor_load_to_lds)
#define HAVE_TDM 1
#else
#define HAVE_TDM 0
#endif

#if HAVE_TDM
typedef __attribute__((ext_vector_type(4))) unsigned int v4u;
typedef __attribute__((ext_vector_type(8))) int          v8i;
typedef __attribute__((ext_vector_type(4))) int          v4i;

__device__ __forceinline__ void tdm_load_2d(
    unsigned lds_byte, const void* gaddr,
    unsigned tensor_d0, unsigned tensor_d1,
    unsigned tile_d0, unsigned tile_d1, unsigned d0_stride) {
  const unsigned long long ga = (unsigned long long)gaddr;
  v4u g0;
  g0[0] = 1u;                                   // count=1 (valid), user mode
  g0[1] = lds_byte;                             // lds_addr (bytes)
  g0[2] = (unsigned)(ga & 0xffffffffu);         // global_addr[31:0]
  g0[3] = (unsigned)((ga >> 32) & 0x01ffffffu)  // global_addr[56:32]
        | (2u << 30);                           // type=2 ("image")
  v8i g1;
  g1[0] = (int)((1u << 16)                      // data_size = 2 bytes
              | (1u << 20)                      // pad_enable
              | (3u << 22)                      // pad_interval: 16 DWORDs
              | (3u << 25));                    // pad_amount:   4 DWORDs
  g1[1] = (int)((tensor_d0 & 0xffffu) << 16);                      // dim0[15:0]
  g1[2] = (int)(((tensor_d0 >> 16) & 0xffffu)
              | ((tensor_d1 & 0xffffu) << 16));                    // dim0 hi, dim1 lo
  g1[3] = (int)(((tensor_d1 >> 16) & 0xffffu)
              | ((tile_d0 & 0xffffu) << 16));                      // dim1 hi, tile_dim0
  g1[4] = (int)(tile_d1 & 0xffffu);                                // tile_dim1 (tile_dim2=0)
  g1[5] = (int)d0_stride;                                          // dim0 stride lo32
  g1[6] = 0;                                                       // stride hi / dim1_stride lo
  g1[7] = 0;
  v4i z4 = {};
  v8i z8 = {};
  __builtin_amdgcn_tensor_load_to_lds(g0, g1, z4, z4, z8, 0);
}
#endif

// ---------------------------------------------------------------------------
// Fragment load from LDS.
// 16-bit A-matrix 16x32 layout (ISA 7.12.2): lanes 0-15 hold row M=lane,
// K={0..7,16..23}; lanes 16-31 hold row M=lane-16, K={8..15,24..31}.
// B (32x16) mirrors this with the tile stored column-major (i.e. Bt[n][k]).
// `ld` must be a multiple of 8 halves so the v8h loads are 16B aligned.
// ---------------------------------------------------------------------------
__device__ __forceinline__ v16h frag_ld(const _Float16* base, int ld) {
  const int lane = threadIdx.x & 31;
  const int r  = lane & 15;
  const int s  = (lane >> 4) << 3;   // 0 or 8
  const _Float16* p = base + r * ld + s;
  v8h lo = *(const v8h*)(p);
  v8h hi = *(const v8h*)(p + 16);
  return __builtin_shufflevector(lo, hi, 0,1,2,3,4,5,6,7,8,9,10,11,12,13,14,15);
}

__device__ __forceinline__ v8f wmma_f16(v16h a, v16h b, v8f c) {
  return __builtin_amdgcn_wmma_f32_16x16x32_f16(
      /*neg_a=*/false, a, /*neg_b=*/false, b,
      /*c_mod=*/(short)0, c, /*reuse_a=*/false, /*reuse_b=*/false);
}

// ---------------------------------------------------------------------------
// Prep kernels: f32 -> f16 conversions
// ---------------------------------------------------------------------------
__global__ __launch_bounds__(256) void cvt_f16_kernel(const float* __restrict__ x,
                                                      _Float16* __restrict__ y, long n) {
  long i = (long)blockIdx.x * blockDim.x + threadIdx.x;
  long stride = (long)gridDim.x * blockDim.x;
  for (; i < n; i += stride) y[i] = (_Float16)x[i];
}

__global__ __launch_bounds__(256) void avg_cvt_kernel(const float* __restrict__ a,
                                                      const float* __restrict__ b,
                                                      _Float16* __restrict__ y, long n) {
  long i = (long)blockIdx.x * blockDim.x + threadIdx.x;
  long stride = (long)gridDim.x * blockDim.x;
  for (; i < n; i += stride) y[i] = (_Float16)((a[i] + b[i]) * 0.5f);
}

// W (K x N, row major f32)  ->  Wt (N x K, row major f16)
__global__ __launch_bounds__(256) void transpose_cvt_kernel(const float* __restrict__ W,
                                                            _Float16* __restrict__ Wt) {
  __shared__ float tile[32][33];
  const int bx = blockIdx.x * 32;   // n
  const int by = blockIdx.y * 32;   // k
  const int tx = threadIdx.x & 31;
  const int ty = threadIdx.x >> 5;  // 0..7
  #pragma unroll
  for (int i = ty; i < 32; i += 8)
    tile[i][tx] = W[(long)(by + i) * HD + bx + tx];
  __syncthreads();
  #pragma unroll
  for (int i = ty; i < 32; i += 8)
    Wt[(long)(bx + i) * HD + by + tx] = (_Float16)tile[tx][i];
}

// ---------------------------------------------------------------------------
// WMMA GEMM:  C[M x N] = A[M x K] @ Bt[N x K]^T + bias
// Block tile 128x128, BK=32, double-buffered LDS. A tile staged with
// per-lane GLOBAL_LOAD_ASYNC_TO_LDS_B128 (row-remapped addressing);
// B (weight) tile staged by the Tensor Data Mover (one tensor_load_to_lds
// per tile, wave 0 issues, TENSORcnt-tracked). 256 threads = 8 waves in a
// 4(M) x 2(N) grid; each wave owns a 32x64 output = 8 WMMA accumulators.
// Row remap: logical row r reads  A row  (r/Ri)*Si + (r%Ri) + in_off
//            and writes       C row  (r/Ri)*So + (r%Ri) + out_off
// ---------------------------------------------------------------------------
#define BM 128
#define BN 128
#define BK 32
#define LDA 40   // BK + 8 pad, multiple of 8 halves (16B aligned rows)

__global__ __launch_bounds__(256) void gemm_wmma_kernel(
    const _Float16* __restrict__ A, const _Float16* __restrict__ Bt,
    const float* __restrict__ bias, _Float16* __restrict__ Cf16,
    float* __restrict__ Cf32, int N, int K,
    int Ri, int Si, int in_off, int So, int out_off) {
  __shared__ __align__(16) _Float16 sA[2][BM * LDA];
  __shared__ __align__(16) _Float16 sB[2][BN * LDA];

  const int tid  = threadIdx.x;
  const int wave = tid >> 5;
  const int wm   = wave >> 1;       // 0..3 (row group of 32)
  const int wn   = wave & 1;        // 0..1 (col group of 64)
  const int bn   = blockIdx.x * BN;
  const int bm   = blockIdx.y * BM;

  // global->LDS mapping: each 128x32 tile = 512 v8h chunks = 2 per thread.
  const int lr = tid >> 2;             // 0..63 (chunk rows lr and lr+64)
  const int lc = (tid & 3) << 3;       // 0,8,16,24
  const int g0 = bm + lr;
  const int g1 = bm + lr + 64;
  const long ar0 = (long)(g0 / Ri) * Si + (g0 % Ri) + in_off;
  const long ar1 = (long)(g1 / Ri) * Si + (g1 % Ri) + in_off;
  const _Float16* Ap0 = A  + ar0 * (long)K + lc;
  const _Float16* Ap1 = A  + ar1 * (long)K + lc;
  const _Float16* Bp0 = Bt + (long)(bn + lr) * K + lc;
  const _Float16* Bp1 = Bt + (long)(bn + lr + 64) * K + lc;
  (void)Bp0; (void)Bp1;

  const unsigned bufStride = (unsigned)(BM * LDA * 2);
  const unsigned offA = lds_off(&sA[0][0]) + (unsigned)((lr * LDA + lc) * 2);
  const unsigned offB = lds_off(&sB[0][0]) + (unsigned)((lr * LDA + lc) * 2);
  const unsigned sB0  = lds_off(&sB[0][0]);
  const unsigned rowHop = (unsigned)(64 * LDA * 2);
  (void)offB; (void)sB0;
#if HAVE_TDM
  const _Float16* Bbase = Bt + (long)bn * K;
#endif

  v8f acc[2][4];
  #pragma unroll
  for (int i = 0; i < 2; ++i)
    #pragma unroll
    for (int j = 0; j < 4; ++j) acc[i][j] = {};

  // Prologue: stage tile 0 into buffer 0.
  async_ld_b128(offA, Ap0);
  async_ld_b128(offA + rowHop, Ap1);
#if HAVE_TDM
  if (wave == 0)
    tdm_load_2d(sB0, Bbase, (unsigned)K, (unsigned)N, BK, BN, (unsigned)K);
#else
  async_ld_b128(offB, Bp0);
  async_ld_b128(offB + rowHop, Bp1);
#endif

  int buf = 0;
  for (int k0 = 0; k0 < K; k0 += BK) {
    const bool hasNext = (k0 + BK) < K;
    if (hasNext) {
      const unsigned nb = (buf ^ 1) ? bufStride : 0u;
      async_ld_b128(offA + nb, Ap0 + k0 + BK);
      async_ld_b128(offA + nb + rowHop, Ap1 + k0 + BK);
#if HAVE_TDM
      if (wave == 0)
        tdm_load_2d(sB0 + nb, Bbase + k0 + BK, (unsigned)K, (unsigned)N,
                    BK, BN, (unsigned)K);
      wait_async_le2();     // this tile's A loads retired, next tile in flight
      if (wave == 0) __builtin_amdgcn_s_wait_tensorcnt(1);  // this tile's TDM done
#else
      async_ld_b128(offB + nb, Bp0 + k0 + BK);
      async_ld_b128(offB + nb + rowHop, Bp1 + k0 + BK);
      wait_async_le4();
#endif
    } else {
      wait_async_0();
#if HAVE_TDM
      if (wave == 0) __builtin_amdgcn_s_wait_tensorcnt(0);
#endif
    }
    __syncthreads();

    v16h a0 = frag_ld(&sA[buf][(wm * 32) * LDA], LDA);
    v16h a1 = frag_ld(&sA[buf][(wm * 32 + 16) * LDA], LDA);
    #pragma unroll
    for (int j = 0; j < 4; ++j) {
      v16h b = frag_ld(&sB[buf][(wn * 64 + j * 16) * LDA], LDA);
      acc[0][j] = wmma_f16(a0, b, acc[0][j]);
      acc[1][j] = wmma_f16(a1, b, acc[1][j]);
    }
    __syncthreads();
    buf ^= 1;
  }

  // Epilogue. C/D layout (ISA 7.12.2): VGPR v, lanes 0-15 -> M=v, N=lane;
  // lanes 16-31 -> M=v+8, N=lane-16.
  const int lane = tid & 31;
  const int hs   = lane >> 4;
  const int cn   = lane & 15;
  #pragma unroll
  for (int i = 0; i < 2; ++i) {
    #pragma unroll
    for (int v = 0; v < 8; ++v) {
      const int r = bm + wm * 32 + i * 16 + v + hs * 8;
      const long orow = (long)(r / Ri) * So + (r % Ri) + out_off;
      #pragma unroll
      for (int j = 0; j < 4; ++j) {
        const int c = bn + wn * 64 + j * 16 + cn;
        const float x = acc[i][j][v] + bias[c];
        if (Cf16) Cf16[orow * (long)N + c] = (_Float16)x;
        else      Cf32[orow * (long)N + c] = x;
      }
    }
  }
}

// ---------------------------------------------------------------------------
// Flash attention: one block = (batch b, head h, 64-query tile).
// Key loop in tiles of 64; S = Q K^T via WMMA, online softmax through LDS,
// O accumulated with WMMA P @ V (V staged transposed in LDS).
// Q and K staged with async global->LDS; next key tile prefetched.
// ---------------------------------------------------------------------------
#define LDH 72   // DHEAD + 8 pad

__global__ __launch_bounds__(256) void attn_kernel(
    const _Float16* __restrict__ Q, const _Float16* __restrict__ Km,
    const _Float16* __restrict__ Vm, const float* __restrict__ mask,
    _Float16* __restrict__ ctx) {
  __shared__ __align__(16) _Float16 sQ [64 * LDH];
  __shared__ __align__(16) _Float16 sK [64 * LDH];
  __shared__ __align__(16) _Float16 sVt[64 * LDH];
  __shared__ __align__(16) _Float16 sP [64 * LDH];
  __shared__ float    sS [64 * 65];
  __shared__ float    s_c[64];

  const int tid  = threadIdx.x;
  const int wave = tid >> 5;
  const int wm   = wave >> 1;
  const int wn   = wave & 1;
  const int lane = tid & 31;
  const int hs   = lane >> 4;
  const int cn   = lane & 15;

  const int qt = blockIdx.x;          // query tile (of 18)
  const int h  = blockIdx.y;          // head
  const int bb = blockIdx.z;          // batch
  const long rowbase = (long)bb * TT + qt * 64;
  const int  hoff = h * DHEAD;

  // per-thread tile chunk mapping (64x64 halves = 512 v8h chunks, 2/thread)
  const int cr0 = tid >> 3;            // rows cr0 and cr0+32
  const int cc  = (tid & 7) << 3;

  // Stage Q tile asynchronously (waited at first loop barrier).
  async_ld_b128(lds_off(&sQ[cr0 * LDH + cc]),
                Q + (rowbase + cr0) * HD + hoff + cc);
  async_ld_b128(lds_off(&sQ[(cr0 + 32) * LDH + cc]),
                Q + (rowbase + cr0 + 32) * HD + hoff + cc);

  v8f o0 = {};
  v8f o1 = {};
  float m_prev = -1e30f;
  float l_run  = 0.0f;

  for (int kt = 0; kt < TT; kt += 64) {
    // Stage K via async; V via sync load + transposed LDS scatter.
    const long kvrow = (long)bb * TT + kt;
    async_ld_b128(lds_off(&sK[cr0 * LDH + cc]),
                  Km + (kvrow + cr0) * HD + hoff + cc);
    async_ld_b128(lds_off(&sK[(cr0 + 32) * LDH + cc]),
                  Km + (kvrow + cr0 + 32) * HD + hoff + cc);
    #pragma unroll
    for (int it = 0; it < 2; ++it) {
      const int r = cr0 + it * 32;
      v8h vv = *(const v8h*)(Vm + (kvrow + r) * HD + hoff + cc);
      #pragma unroll
      for (int j = 0; j < 8; ++j) sVt[(cc + j) * LDH + r] = vv[j];
    }
    if (kt + 64 < TT) {   // prefetch next key tile (global_prefetch_b8)
      __builtin_prefetch(Km + (kvrow + 64 + cr0) * HD + hoff + cc, 0, 1);
      __builtin_prefetch(Vm + (kvrow + 64 + cr0) * HD + hoff + cc, 0, 1);
    }
    wait_async_0();
    __syncthreads();

    // S tile = Q @ K^T  (64 x 64), per wave: 16 rows x 32 keys, 2 K-chunks.
    v8f s0 = {};
    v8f s1 = {};
    #pragma unroll
    for (int kc = 0; kc < 2; ++kc) {
      v16h a  = frag_ld(sQ + (wm * 16) * LDH + kc * 32, LDH);
      v16h b0 = frag_ld(sK + (wn * 32) * LDH + kc * 32, LDH);
      v16h b1 = frag_ld(sK + (wn * 32 + 16) * LDH + kc * 32, LDH);
      s0 = wmma_f16(a, b0, s0);
      s1 = wmma_f16(a, b1, s1);
    }
    #pragma unroll
    for (int v = 0; v < 8; ++v) {
      const int r  = wm * 16 + v + hs * 8;
      const int c0 = wn * 32 + cn;
      const int c1 = c0 + 16;
      sS[r * 65 + c0] = s0[v] * 0.125f + mask[(long)bb * TT + kt + c0];
      sS[r * 65 + c1] = s1[v] * 0.125f + mask[(long)bb * TT + kt + c1];
    }
    __syncthreads();

    // Online softmax: lane t owns row t.
    if (tid < 64) {
      float mx = m_prev;
      #pragma unroll 8
      for (int c = 0; c < 64; ++c) mx = fmaxf(mx, sS[tid * 65 + c]);
      const float corr = __expf(m_prev - mx);
      float sum = 0.0f;
      #pragma unroll 8
      for (int c = 0; c < 64; ++c) {
        const float p = __expf(sS[tid * 65 + c] - mx);
        sP[tid * LDH + c] = (_Float16)p;
        sum += p;
      }
      l_run  = l_run * corr + sum;
      m_prev = mx;
      s_c[tid] = corr;
    }
    __syncthreads();

    // Rescale running O by the per-row correction, then O += P @ V.
    #pragma unroll
    for (int v = 0; v < 8; ++v) {
      const float corr = s_c[wm * 16 + v + hs * 8];
      o0[v] *= corr;
      o1[v] *= corr;
    }
    #pragma unroll
    for (int kc = 0; kc < 2; ++kc) {
      v16h a  = frag_ld(sP  + (wm * 16) * LDH + kc * 32, LDH);
      v16h b0 = frag_ld(sVt + (wn * 32) * LDH + kc * 32, LDH);
      v16h b1 = frag_ld(sVt + (wn * 32 + 16) * LDH + kc * 32, LDH);
      o0 = wmma_f16(a, b0, o0);
      o1 = wmma_f16(a, b1, o1);
    }
    __syncthreads();
  }

  if (tid < 64) s_c[tid] = 1.0f / l_run;
  __syncthreads();

  #pragma unroll
  for (int v = 0; v < 8; ++v) {
    const int r = wm * 16 + v + hs * 8;
    const float inv = s_c[r];
    const int c0 = wn * 32 + cn;
    const int c1 = c0 + 16;
    ctx[(rowbase + r) * HD + hoff + c0] = (_Float16)(o0[v] * inv);
    ctx[(rowbase + r) * HD + hoff + c1] = (_Float16)(o1[v] * inv);
  }
}

// ---------------------------------------------------------------------------
// Residual + LayerNorm: one block per row of H=1024 (4 elems/thread).
// (GEMM epilogue already added the projection bias.)
// ---------------------------------------------------------------------------
__global__ __launch_bounds__(256) void ln_kernel(
    const float* __restrict__ pre, const float* __restrict__ resid,
    const float* __restrict__ g, const float* __restrict__ bvec,
    float* __restrict__ out) {
  __shared__ float red[256];
  __shared__ float s_mu, s_rstd;
  const long row = blockIdx.x;
  const int  tid = threadIdx.x;
  float x[4];
  float sum = 0.0f;
  #pragma unroll
  for (int i = 0; i < 4; ++i) {
    const int c = tid + i * 256;
    x[i] = pre[row * HD + c] + resid[row * HD + c];
    sum += x[i];
  }
  red[tid] = sum;
  __syncthreads();
  for (int s = 128; s > 0; s >>= 1) {
    if (tid < s) red[tid] += red[tid + s];
    __syncthreads();
  }
  if (tid == 0) s_mu = red[0] * (1.0f / HD);
  __syncthreads();
  const float mu = s_mu;
  float vs = 0.0f;
  #pragma unroll
  for (int i = 0; i < 4; ++i) {
    const float d = x[i] - mu;
    vs += d * d;
  }
  red[tid] = vs;
  __syncthreads();
  for (int s = 128; s > 0; s >>= 1) {
    if (tid < s) red[tid] += red[tid + s];
    __syncthreads();
  }
  if (tid == 0) s_rstd = rsqrtf(red[0] * (1.0f / HD) + 1e-12f);
  __syncthreads();
  const float rstd = s_rstd;
  #pragma unroll
  for (int i = 0; i < 4; ++i) {
    const int c = tid + i * 256;
    out[row * HD + c] = (x[i] - mu) * rstd * g[c] + bvec[c];
  }
}

// ---------------------------------------------------------------------------
// Host launcher
// ---------------------------------------------------------------------------
extern "C" void kernel_launch(void* const* d_in, const int* in_sizes, int n_in,
                              void* d_out, int out_size, void* d_ws, size_t ws_size,
                              hipStream_t stream) {
  (void)in_sizes; (void)n_in; (void)out_size; (void)ws_size;

  const float* word  = (const float*)d_in[0];   // B x TW x H
  const float* ent   = (const float*)d_in[1];   // B x TE x H
  const float* mask  = (const float*)d_in[2];   // B x 1 x 1 x T
  const float* qpos  = (const float*)d_in[3];   // B x TE x H
  const float* Wmat[8] = { (const float*)d_in[4],  (const float*)d_in[6],
                           (const float*)d_in[8],  (const float*)d_in[10],
                           (const float*)d_in[12], (const float*)d_in[14],
                           (const float*)d_in[16], (const float*)d_in[18] };
  // order: Wq, Wk, Wv, Weq, Wek, Wev, Wo, Weo
  const float* bq  = (const float*)d_in[5];
  const float* bk  = (const float*)d_in[7];
  const float* bv  = (const float*)d_in[9];
  const float* beq = (const float*)d_in[11];
  const float* bek = (const float*)d_in[13];
  const float* bev = (const float*)d_in[15];
  const float* bo  = (const float*)d_in[17];
  const float* beo = (const float*)d_in[19];
  const float* ln_g  = (const float*)d_in[20];
  const float* ln_b  = (const float*)d_in[21];
  const float* eln_g = (const float*)d_in[22];
  const float* eln_b = (const float*)d_in[23];

  const long nWord = (long)BATCH * TW * HD;   // 4,194,304
  const long nEnt  = (long)BATCH * TE * HD;   //   524,288
  const long nQKV  = (long)BATCH * TT * HD;   // 4,718,592

  // Workspace carving (all 256B aligned by construction).
  char* p = (char*)d_ws;
  _Float16* word16 = (_Float16*)p;  p += nWord * 2;
  _Float16* ent16  = (_Float16*)p;  p += nEnt  * 2;
  _Float16* pos16  = (_Float16*)p;  p += nEnt  * 2;
  _Float16* Wt[8];
  for (int i = 0; i < 8; ++i) { Wt[i] = (_Float16*)p; p += (long)HD * HD * 2; }
  _Float16* Qb   = (_Float16*)p;  p += nQKV * 2;
  _Float16* Kb   = (_Float16*)p;  p += nQKV * 2;
  _Float16* Vb   = (_Float16*)p;  p += nQKV * 2;
  _Float16* Ctx  = (_Float16*)p;  p += nQKV * 2;
  float* wordPre = (float*)p;     p += nWord * 4;
  float* entPre  = (float*)p;     p += nEnt  * 4;

  float* outWord = (float*)d_out;
  float* outEnt  = outWord + nWord;

  // --- Prep: conversions -------------------------------------------------
  cvt_f16_kernel<<<4096, 256, 0, stream>>>(word, word16, nWord);
  cvt_f16_kernel<<<1024, 256, 0, stream>>>(ent, ent16, nEnt);
  avg_cvt_kernel<<<1024, 256, 0, stream>>>(ent, qpos, pos16, nEnt);
  dim3 tgrid(HD / 32, HD / 32);
  for (int i = 0; i < 8; ++i)
    transpose_cvt_kernel<<<tgrid, 256, 0, stream>>>(Wmat[i], Wt[i]);

  // --- Projections (write into concatenated B x T x H Q/K/V buffers) -----
  dim3 gw(HD / BN, (BATCH * TW) / BM);   // (8, 32)
  dim3 ge(HD / BN, (BATCH * TE) / BM);   // (8, 4)
  // word: Ri=TW, Si=TW (contiguous in), So=TT, out_off=0
  gemm_wmma_kernel<<<gw, 256, 0, stream>>>(word16, Wt[0], bq, Qb, nullptr,
                                           HD, HD, TW, TW, 0, TT, 0);
  gemm_wmma_kernel<<<gw, 256, 0, stream>>>(word16, Wt[1], bk, Kb, nullptr,
                                           HD, HD, TW, TW, 0, TT, 0);
  gemm_wmma_kernel<<<gw, 256, 0, stream>>>(word16, Wt[2], bv, Vb, nullptr,
                                           HD, HD, TW, TW, 0, TT, 0);
  // entity: Ri=TE, Si=TE, So=TT, out_off=TW
  gemm_wmma_kernel<<<ge, 256, 0, stream>>>(pos16, Wt[3], beq, Qb, nullptr,
                                           HD, HD, TE, TE, 0, TT, TW);
  gemm_wmma_kernel<<<ge, 256, 0, stream>>>(pos16, Wt[4], bek, Kb, nullptr,
                                           HD, HD, TE, TE, 0, TT, TW);
  gemm_wmma_kernel<<<ge, 256, 0, stream>>>(ent16, Wt[5], bev, Vb, nullptr,
                                           HD, HD, TE, TE, 0, TT, TW);

  // --- Attention ----------------------------------------------------------
  dim3 agrid(TT / 64, NHEAD, BATCH);
  attn_kernel<<<agrid, 256, 0, stream>>>(Qb, Kb, Vb, mask, Ctx);

  // --- Output projections (read remapped ctx rows, f32 out + bias) --------
  gemm_wmma_kernel<<<gw, 256, 0, stream>>>(Ctx, Wt[6], bo, nullptr, wordPre,
                                           HD, HD, TW, TT, 0, TW, 0);
  gemm_wmma_kernel<<<ge, 256, 0, stream>>>(Ctx, Wt[7], beo, nullptr, entPre,
                                           HD, HD, TE, TT, TW, TE, 0);

  // --- Residual + LayerNorm ----------------------------------------------
  ln_kernel<<<BATCH * TW, 256, 0, stream>>>(wordPre, word, ln_g, ln_b, outWord);
  ln_kernel<<<BATCH * TE, 256, 0, stream>>>(entPre, ent, eln_g, eln_b, outEnt);
}
